// Attention_47622597378289
// MI455X (gfx1250) — matrix operand order repaired and checked by reading, not compiled, not run on hardware
//
#include <hip/hip_runtime.h>

// ---------------------------------------------------------------------------
// Two cascaded attention blocks on MI455X (gfx1250, wave32, WMMA bf16).
//   B=8, C=512, C8=64, H=W=48, N=2304
// Big GEMMs (scores N x N x 64, out N x N x 512) run on
// v_wmma_f32_16x16x32_bf16. Positional bias folded into K at projection:
// q@k + q@pos == q@(k+pos).
// ---------------------------------------------------------------------------

#define BATCH 8
#define CH    512
#define C8    64
#define HH    48
#define WW    48
#define NN    2304            // H*W
#define NT    144             // NN/16
#define NST   36              // NN/64 (out-kernel n super-tiles)
#define NPAD  2308            // padded row length for LDS softmax buffer

typedef __attribute__((ext_vector_type(16))) __bf16        v16bf;
typedef __attribute__((ext_vector_type(8)))  float         v8f;
typedef __attribute__((ext_vector_type(4)))  unsigned int  uint4v;

// round-to-nearest-even f32 -> bf16 (stored as raw u16)
static __device__ __forceinline__ unsigned short f2bf(float f) {
    unsigned u = __float_as_uint(f);
    unsigned r = u + 0x7FFFu + ((u >> 16) & 1u);
    return (unsigned short)(r >> 16);
}

// Load a 16-element bf16 fragment as two 16B chunks (both 16B aligned).
static __device__ __forceinline__ v16bf load_frag_pair(const unsigned short* p0,
                                                       const unsigned short* p1) {
    union { v16bf v; uint4v u[2]; } f;
    f.u[0] = *reinterpret_cast<const uint4v*>(p0);
    f.u[1] = *reinterpret_cast<const uint4v*>(p1);
    return f.v;
}

// ---------------------------------------------------------------------------
// Projection: out[o][n] (or [n][o]) = sum_c W[o][c] * x[b][c][n] + bias[o]
//   (+ optional positional table ht[o][h] + wt[o][w], folded into K).
// Tiled LDS f32 GEMM, tile = 64 o x 16 n, 256 threads (4 outputs/thread).
// layout 0: out[(b*N + n)*O + o]  (bf16, [N][O] -- WMMA A/B friendly)
// layout 1: out[(b*O + o)*N + n]  (bf16, [O][N] -- WMMA A for att@v^T)
// ---------------------------------------------------------------------------
__global__ void proj_kernel(const float* __restrict__ x,
                            const float* __restrict__ W,
                            const float* __restrict__ bias,
                            const float* __restrict__ ht,   // may be nullptr
                            const float* __restrict__ wt,   // may be nullptr
                            unsigned short* __restrict__ out,
                            int O_total, int numOt, int layout) {
    __shared__ float xs[64][17];
    __shared__ float ws[64][65];

    int bx    = blockIdx.x;
    int ntile = bx % NT;
    int tmp   = bx / NT;
    int otile = tmp % numOt;
    int b     = tmp / numOt;
    int n0    = ntile * 16;

    int tid = threadIdx.x;
    int o   = tid & 63;            // local o
    int ng  = tid >> 6;            // 0..3, n-subgroup
    int o_g = otile * 64 + o;

    float acc[4] = {0.f, 0.f, 0.f, 0.f};

    for (int cc0 = 0; cc0 < CH; cc0 += 64) {
        // stage x tile [64 c x 16 n] (coalesced along n)
        #pragma unroll
        for (int p = 0; p < 4; ++p) {
            int ci = (tid >> 4) + 16 * p;
            int nj = tid & 15;
            xs[ci][nj] = x[((size_t)(b * CH + cc0 + ci)) * NN + n0 + nj];
        }
        // stage W tile [64 o x 64 c] (coalesced along c)
        #pragma unroll
        for (int p = 0; p < 16; ++p) {
            int oi  = (tid >> 6) + 4 * p;
            int ccj = tid & 63;
            ws[oi][ccj] = W[((size_t)(otile * 64 + oi)) * CH + cc0 + ccj];
        }
        __syncthreads();
        #pragma unroll 8
        for (int cc = 0; cc < 64; ++cc) {
            float wv = ws[o][cc];
            #pragma unroll
            for (int t = 0; t < 4; ++t)
                acc[t] += wv * xs[cc][ng * 4 + t];
        }
        __syncthreads();
    }

    float bval = bias[o_g];
    #pragma unroll
    for (int t = 0; t < 4; ++t) {
        int n = n0 + ng * 4 + t;
        float r = acc[t] + bval;
        if (ht != nullptr) {   // fold positional bias into K
            int h = n / WW, w = n % WW;
            r += ht[o_g * HH + h] + wt[o_g * WW + w];
        }
        if (layout == 0)
            out[((size_t)b * NN + n) * O_total + o_g] = f2bf(r);
        else
            out[((size_t)b * O_total + o_g) * NN + n] = f2bf(r);
    }
}

// ---------------------------------------------------------------------------
// Scores + softmax: S[n][m] = sum_k q[n][k] * kp[m][k], row softmax -> att bf16.
// One workgroup (8 waves) per (b, 16-query tile). Each wave holds the 2 Q
// A-frags in VGPRs and sweeps 18 of the 144 m-tiles, TWO per iteration with
// independent accumulator chains (interleaved WMMAs cover the D->ds_store
// hazard). C tiles land in a 16 x NPAD LDS row buffer (148 KB; 320 KB WGP LDS
// -> 2 workgroups per WGP).
// ---------------------------------------------------------------------------
__global__ void scores_kernel(const unsigned short* __restrict__ q,
                              const unsigned short* __restrict__ kp,
                              unsigned short* __restrict__ att) {
    extern __shared__ float s_buf[];   // [16][NPAD]

    int bx    = blockIdx.x;
    int ntile = bx % NT;
    int b     = bx / NT;
    int n0    = ntile * 16;

    int tid  = threadIdx.x;
    int wave = tid >> 5;
    int lane = tid & 31;

    // A fragments: row = lane&15, K halves per ISA layout
    int rowA  = lane & 15;
    int kbase = (lane & 16) ? 8 : 0;
    const unsigned short* qrow = q + ((size_t)(b * NN + n0 + rowA)) * C8;
    v16bf a0 = load_frag_pair(qrow + kbase,      qrow + kbase + 16);       // K 0..31
    v16bf a1 = load_frag_pair(qrow + 32 + kbase, qrow + 32 + kbase + 16);  // K 32..63

    int col   = lane & 15;
    int kb2   = (lane & 16) ? 8 : 0;
    int rbase = (lane & 16) ? 8 : 0;

    // 144 m-tiles; each wave takes tile pairs {wave*2, wave*2+1} + 16*i.
    for (int base = wave * 2; base < NT; base += 16) {
        int m0a = base * 16;
        int m0b = m0a + 16;
        // B fragment: B[k][m] = kp[m][k] -> two 16B chunks per lane (K halves per ISA layout)
        const unsigned short* kpa = kp + ((size_t)(b * NN + m0a + col)) * C8;
        const unsigned short* kpb2p = kp + ((size_t)(b * NN + m0b + col)) * C8;
        v16bf b0a = load_frag_pair(kpa + kb2,        kpa + kb2 + 16);
        v16bf b1a = load_frag_pair(kpa + 32 + kb2,   kpa + 32 + kb2 + 16);
        v16bf b0b = load_frag_pair(kpb2p + kb2,      kpb2p + kb2 + 16);
        v16bf b1b = load_frag_pair(kpb2p + 32 + kb2, kpb2p + 32 + kb2 + 16);

        v8f ca = {}, cb = {};
        ca = __builtin_amdgcn_wmma_f32_16x16x32_bf16(false, a0, false, b0a,
                                                     (short)0, ca, false, false);
        cb = __builtin_amdgcn_wmma_f32_16x16x32_bf16(false, a0, false, b0b,
                                                     (short)0, cb, false, false);
        ca = __builtin_amdgcn_wmma_f32_16x16x32_bf16(false, a1, false, b1a,
                                                     (short)0, ca, false, false);
        cb = __builtin_amdgcn_wmma_f32_16x16x32_bf16(false, a1, false, b1b,
                                                     (short)0, cb, false, false);
        #pragma unroll
        for (int r = 0; r < 8; ++r)
            s_buf[(size_t)(rbase + r) * NPAD + m0a + col] = ca[r];
        #pragma unroll
        for (int r = 0; r < 8; ++r)
            s_buf[(size_t)(rbase + r) * NPAD + m0b + col] = cb[r];
    }
    __syncthreads();

    // Softmax: each wave owns 2 of the 16 rows; shuffle reductions (wave32).
    #pragma unroll
    for (int rr = 0; rr < 2; ++rr) {
        int row = wave * 2 + rr;
        float* srow = s_buf + (size_t)row * NPAD;

        float mx = -3.4e38f;
        for (int i = lane; i < NN; i += 32) mx = fmaxf(mx, srow[i]);
        #pragma unroll
        for (int off = 16; off > 0; off >>= 1)
            mx = fmaxf(mx, __shfl_xor(mx, off, 32));

        float sum = 0.f;
        for (int i = lane; i < NN; i += 32) {
            float e = __expf(srow[i] - mx);
            srow[i] = e;
            sum += e;
        }
        #pragma unroll
        for (int off = 16; off > 0; off >>= 1)
            sum += __shfl_xor(sum, off, 32);
        float inv = 1.0f / sum;

        unsigned short* arow = att + ((size_t)(b * NN + n0 + row)) * NN;
        for (int i = lane; i < NN; i += 32)
            arow[i] = f2bf(srow[i] * inv);
    }
}

// ---------------------------------------------------------------------------
// Output GEMM + epilogue: dest[b][c][n] = gamma * sum_m v[c][m]*att[n][m] + res.
// One wave per 16c x 64n tile: 4 INDEPENDENT accumulators -> 4x reuse of the
// v A-fragment and WMMA dependency distance 4 (hides XDL latency).
// K loop steps 64 (8 WMMAs / iter from 4x16B + 8x32B loads).
// A = v[c][m] (row-contiguous), B[k=m][n] = att[n][m] (two 16B chunks/lane).
// ---------------------------------------------------------------------------
__global__ void out_kernel(const unsigned short* __restrict__ att,
                           const unsigned short* __restrict__ v,
                           const float* __restrict__ residual,
                           const float* __restrict__ gamma,
                           float* __restrict__ dest) {
    int bx     = blockIdx.x;
    int cgroup = bx % 4;
    int tmp    = bx / 4;
    int nst    = tmp % NST;          // 64-wide n super-tile
    int b      = tmp / NST;
    int n0     = nst * 64;

    int tid  = threadIdx.x;
    int wave = tid >> 5;
    int lane = tid & 31;
    int c0   = (cgroup * 8 + wave) * 16;

    int rowA  = lane & 15;
    int kbase = (lane & 16) ? 8 : 0;
    int col   = lane & 15;
    int kb2   = (lane & 16) ? 8 : 0;

    const unsigned short* vrow = v + ((size_t)(b * CH + c0 + rowA)) * NN;
    const unsigned short* arow[4];
    #pragma unroll
    for (int j = 0; j < 4; ++j)
        arow[j] = att + ((size_t)(b * NN + n0 + j * 16 + col)) * NN;

    v8f acc[4] = {v8f{}, v8f{}, v8f{}, v8f{}};

    for (int m0 = 0; m0 < NN; m0 += 64) {
        if (m0 + 512 < NN) {
            __builtin_prefetch(vrow + m0 + 512, 0, 0);
            __builtin_prefetch(arow[0] + m0 + 512, 0, 0);
            __builtin_prefetch(arow[1] + m0 + 512, 0, 0);
            __builtin_prefetch(arow[2] + m0 + 512, 0, 0);
            __builtin_prefetch(arow[3] + m0 + 512, 0, 0);
        }
        // A fragments (shared by all 4 n-subtiles): K m0..m0+31 and m0+32..m0+63
        v16bf av0 = load_frag_pair(vrow + m0 + kbase,      vrow + m0 + kbase + 16);
        v16bf av1 = load_frag_pair(vrow + m0 + 32 + kbase, vrow + m0 + 32 + kbase + 16);
        // B fragments per n-subtile
        v16bf b0[4], b1[4];
        #pragma unroll
        for (int j = 0; j < 4; ++j) {
            b0[j] = load_frag_pair(arow[j] + m0 + kb2,      arow[j] + m0 + kb2 + 16);
            b1[j] = load_frag_pair(arow[j] + m0 + 32 + kb2, arow[j] + m0 + 32 + kb2 + 16);
        }
        // 4 independent chains, interleaved: dependency distance = 4 WMMAs
        #pragma unroll
        for (int j = 0; j < 4; ++j)
            acc[j] = __builtin_amdgcn_wmma_f32_16x16x32_bf16(false, av0, false, b0[j],
                                                             (short)0, acc[j], false, false);
        #pragma unroll
        for (int j = 0; j < 4; ++j)
            acc[j] = __builtin_amdgcn_wmma_f32_16x16x32_bf16(false, av1, false, b1[j],
                                                             (short)0, acc[j], false, false);
    }

    float g = gamma[0];
    int rbase = (lane & 16) ? 8 : 0;
    #pragma unroll
    for (int j = 0; j < 4; ++j) {
        #pragma unroll
        for (int r = 0; r < 8; ++r) {
            int c = c0 + rbase + r;
            size_t idx = ((size_t)(b * CH + c)) * NN + n0 + j * 16 + col;
            dest[idx] = g * acc[j][r] + residual[idx];
        }
    }
}

// ---------------------------------------------------------------------------
// Host orchestration
// ---------------------------------------------------------------------------
static void run_stage(const float* qsrc, const float* kvsrc,
                      const float* Wq, const float* bq,
                      const float* Wk, const float* bk,
                      const float* Wv, const float* bv,
                      const float* ht, const float* wt, const float* gamma,
                      const float* residual, float* dest,
                      unsigned short* qb, unsigned short* kpb,
                      unsigned short* vb, unsigned short* attb,
                      hipStream_t stream) {
    // q projection: [N][64], no pos
    proj_kernel<<<BATCH * 1 * NT, 256, 0, stream>>>(qsrc, Wq, bq, nullptr, nullptr,
                                                    qb, C8, 1, 0);
    // k projection + positional bias folded in: [N][64]
    proj_kernel<<<BATCH * 1 * NT, 256, 0, stream>>>(kvsrc, Wk, bk, ht, wt,
                                                    kpb, C8, 1, 0);
    // v projection: [512][N]
    proj_kernel<<<BATCH * 8 * NT, 256, 0, stream>>>(kvsrc, Wv, bv, nullptr, nullptr,
                                                    vb, CH, 8, 1);
    // scores + softmax (dynamic LDS: 16 * NPAD floats = 147,712 B)
    scores_kernel<<<BATCH * NT, 256, 16 * NPAD * sizeof(float), stream>>>(qb, kpb, attb);
    // att @ v^T + gamma/residual epilogue (16c x 64n per wave)
    out_kernel<<<BATCH * NST * 4, 256, 0, stream>>>(attb, vb, residual, gamma, dest);
}

extern "C" void kernel_launch(void* const* d_in, const int* in_sizes, int n_in,
                              void* d_out, int out_size, void* d_ws, size_t ws_size,
                              hipStream_t stream) {
    const float* x     = (const float*)d_in[0];
    const float* total = (const float*)d_in[1];
    const float* eWq = (const float*)d_in[2];  const float* ebq = (const float*)d_in[3];
    const float* eWk = (const float*)d_in[4];  const float* ebk = (const float*)d_in[5];
    const float* eWv = (const float*)d_in[6];  const float* ebv = (const float*)d_in[7];
    const float* eht = (const float*)d_in[8];  const float* ewt = (const float*)d_in[9];
    const float* eg  = (const float*)d_in[10];
    const float* dWq = (const float*)d_in[11]; const float* dbq = (const float*)d_in[12];
    const float* dWk = (const float*)d_in[13]; const float* dbk = (const float*)d_in[14];
    const float* dWv = (const float*)d_in[15]; const float* dbv = (const float*)d_in[16];
    const float* dht = (const float*)d_in[17]; const float* dwt = (const float*)d_in[18];
    const float* dg  = (const float*)d_in[19];

    char* ws = (char*)d_ws;
    size_t off = 0;
    unsigned short* qb   = (unsigned short*)(ws + off); off += (size_t)BATCH * NN * C8 * 2;
    unsigned short* kpb  = (unsigned short*)(ws + off); off += (size_t)BATCH * NN * C8 * 2;
    unsigned short* vb   = (unsigned short*)(ws + off); off += (size_t)BATCH * CH * NN * 2;
    unsigned short* attb = (unsigned short*)(ws + off); off += (size_t)BATCH * NN * NN * 2;
    float*          xenc = (float*)         (ws + off); off += (size_t)BATCH * CH * NN * 4;

    // Encoder: q from `total`, k/v from `x`, residual x -> x_encoder (workspace)
    run_stage(total, x, eWq, ebq, eWk, ebk, eWv, ebv, eht, ewt, eg,
              x, xenc, qb, kpb, vb, attb, stream);
    // Decoder: q from `x`, k/v from x_encoder, residual x -> d_out
    run_stage(x, xenc, dWq, dbq, dWk, dbk, dWv, dbv, dht, dwt, dg,
              x, (float*)d_out, qb, kpb, vb, attb, stream);
}
